// MP_Layer_edge_only_16406775071593
// MI455X (gfx1250) — matrix operand order repaired and checked by reading, not compiled, not run on hardware
//
#include <hip/hip_runtime.h>

// B=4, N=256, D=128, E=128, UP=256, SD=128
// rows = B*N*N = 262144 edge rows, 128 wide in, 256 hidden, 128 out.

typedef _Float16 h16v __attribute__((ext_vector_type(16)));
typedef _Float16 h8v  __attribute__((ext_vector_type(8)));
typedef _Float16 h4v  __attribute__((ext_vector_type(4)));
typedef float    f8v  __attribute__((ext_vector_type(8)));
typedef float    f4v  __attribute__((ext_vector_type(4)));

union HFrag { h16v v; h8v h[2]; };

// ---------------------------------------------------------------------------
// Pre-kernel 1: pib[b,n,:] = nodes[b,n,:] @ W1[:128,:] + b1   (the "pi" term, j-indexed)
//               pjw[b,n,:] = nodes[b,n,:] @ W1[128:256,:]     (the "pj" term, i-indexed)
// 1024 blocks x 256 threads; trivial FLOPs, plain fp32.
// ---------------------------------------------------------------------------
__global__ __launch_bounds__(256) void mp_proj(const float* __restrict__ nodes,
                                               const float* __restrict__ W1,
                                               const float* __restrict__ b1,
                                               float* __restrict__ pib,
                                               float* __restrict__ pjw) {
  __shared__ float sn[128];
  const int row = blockIdx.x;            // b*256 + n
  const int h   = threadIdx.x;           // hidden index, 256
  if (h < 128) sn[h] = nodes[(size_t)row * 128 + h];
  __syncthreads();
  float ai = 0.f, aj = 0.f;
  #pragma unroll 8
  for (int d = 0; d < 128; ++d) {
    const float nd = sn[d];
    ai = fmaf(nd, W1[(size_t)d * 256 + h], ai);
    aj = fmaf(nd, W1[(size_t)(d + 128) * 256 + h], aj);
  }
  pib[(size_t)row * 256 + h] = ai + b1[h];
  pjw[(size_t)row * 256 + h] = aj;
}

// ---------------------------------------------------------------------------
// Pre-kernel 2: f16 transposed weights for WMMA B-fragment-friendly layout.
//   w1et[n][k] = (f16) W1[256+k][n]   n<256, k<128
//   w2t [n][k] = (f16) W2[k][n]       n<128, k<256
// ---------------------------------------------------------------------------
__global__ __launch_bounds__(256) void mp_wcvt(const float* __restrict__ W1,
                                               const float* __restrict__ W2,
                                               _Float16* __restrict__ w1et,
                                               _Float16* __restrict__ w2t) {
  const int idx = blockIdx.x * 256 + threadIdx.x;   // 65536 total
  if (idx < 32768) {
    const int n = idx >> 7, k = idx & 127;
    w1et[idx] = (_Float16)W1[(size_t)(256 + k) * 256 + n];
  } else {
    const int j = idx - 32768;
    const int n = j >> 8, k = j & 255;
    w2t[j] = (_Float16)W2[(size_t)k * 128 + n];
  }
}

// ---------------------------------------------------------------------------
// Main fused kernel: per WG, one 64-row edge tile.
// LDS layout (dynamic, 120832 B):
//   phase 1: sW1 [256][136] f16 (69632 B) | sA [64][136] f16 (17408 B) | sH [64][264] f16 (33792 B)
//   phase 2: sW2 [128][264] f16 (67584 B, aliases sW1/sA)              | sH persists
// ---------------------------------------------------------------------------
__global__ __launch_bounds__(256) void mp_main(const float* __restrict__ edges,
                                               const float* __restrict__ mask,
                                               const float* __restrict__ pib,
                                               const float* __restrict__ pjw,
                                               const _Float16* __restrict__ w1et,
                                               const _Float16* __restrict__ w2t,
                                               const float* __restrict__ b2,
                                               float* __restrict__ out) {
  extern __shared__ __align__(16) char smem[];
  _Float16* const sW1 = (_Float16*)smem;               // [256][136]
  _Float16* const sA  = (_Float16*)(smem + 69632);     // [64][136]
  _Float16* const sH  = (_Float16*)(smem + 87040);     // [64][264]
  _Float16* const sW2 = (_Float16*)smem;               // [128][264] (phase 2)

  const int t    = threadIdx.x;
  const int row0 = blockIdx.x * 64;        // global edge row base
  const int bIdx = row0 >> 16;             // batch
  const int p0   = row0 & 65535;
  const int iIdx = p0 >> 8;                // i is constant over a 64-row tile
  const int j0   = p0 & 255;

  // ---- stage W1_e^T into LDS: 4096 h8 chunks, 16 per thread
  #pragma unroll
  for (int pass = 0; pass < 16; ++pass) {
    const int c8 = pass * 256 + t;
    const int n  = c8 >> 4;
    const int k  = (c8 & 15) * 8;
    *(h8v*)(sW1 + n * 136 + k) = *(const h8v*)(w1et + (size_t)c8 * 8);
  }
  // ---- stage edge tile 64x128 fp32 -> f16 LDS
  {
    const float* eg = edges + (size_t)row0 * 128;
    #pragma unroll
    for (int pass = 0; pass < 8; ++pass) {
      const int idx = pass * 1024 + t * 4;
      const f4v v = *(const f4v*)(eg + idx);
      const int r = idx >> 7, c = idx & 127;
      h4v hv;
      hv[0] = (_Float16)v[0]; hv[1] = (_Float16)v[1];
      hv[2] = (_Float16)v[2]; hv[3] = (_Float16)v[3];
      *(h4v*)(sA + r * 136 + c) = hv;
    }
    if (row0 + 64 < 262144) __builtin_prefetch(eg + 64 * 128, 0, 0); // next tile
  }
  __syncthreads();

  const int w    = t >> 5;
  const int lane = t & 31;
  const int hs   = lane >> 4;      // lane-half select (K-group)
  const int ln   = lane & 15;      // M (A) / N (B,C) index
  const int mt   = w & 3;          // wave's 16-row M tile
  const int ntb  = (w >> 2) * 8;   // phase-1 N-tile base (8 tiles of 16)

  // ---- GEMM1: pe(64x256) = A(64x128) x W1e(128x256), f16 in, f32 acc
  f8v acc[8] = {};
  const _Float16* aRow = sA + (mt * 16 + ln) * 136;
  #pragma unroll
  for (int ks = 0; ks < 4; ++ks) {
    HFrag a;
    const int ak = ks * 32 + hs * 8;
    a.h[0] = *(const h8v*)(aRow + ak);        // VGPR 0..3: K = ak..ak+7
    a.h[1] = *(const h8v*)(aRow + ak + 16);   // VGPR 4..7: K = ak+16..ak+23
    #pragma unroll
    for (int q = 0; q < 8; ++q) {
      const int n = (ntb + q) * 16 + ln;
      HFrag bf;
      const _Float16* bp = sW1 + n * 136 + ks * 32 + hs * 8;
      bf.h[0] = *(const h8v*)bp;
      bf.h[1] = *(const h8v*)(bp + 16);
      acc[q] = __builtin_amdgcn_wmma_f32_16x16x32_f16(
          false, a.v, false, bf.v, (short)0, acc[q], false, false);
    }
  }

  // ---- epilogue 1: h = relu(pe + pj[b,i,:] + (pi+b1)[b,j,:]) -> sH (f16)
  {
    const float* pjr = pjw + (size_t)(bIdx * 256 + iIdx) * 256;
    float pjn[8];
    #pragma unroll
    for (int q = 0; q < 8; ++q) pjn[q] = pjr[(ntb + q) * 16 + ln];
    #pragma unroll
    for (int r = 0; r < 8; ++r) {
      const int m = mt * 16 + r + 8 * hs;    // C layout: VGPR r, lane-half -> row
      const float* pir = pib + (size_t)(bIdx * 256 + j0 + m) * 256;
      _Float16* hRow = sH + m * 264;
      #pragma unroll
      for (int q = 0; q < 8; ++q) {
        const int n = (ntb + q) * 16 + ln;
        hRow[n] = (_Float16)fmaxf(acc[q][r] + pjn[q] + pir[n], 0.f);
      }
    }
  }
  __syncthreads();

  // ---- stage W2^T into LDS (reuse phase-1 region)
  #pragma unroll
  for (int pass = 0; pass < 16; ++pass) {
    const int c8 = pass * 256 + t;
    const int n  = c8 >> 5;
    const int k  = (c8 & 31) * 8;
    *(h8v*)(sW2 + n * 264 + k) = *(const h8v*)(w2t + (size_t)c8 * 8);
  }
  __syncthreads();

  // ---- GEMM2: o(64x128) = H(64x256) x W2(256x128)
  f8v acc2[4] = {};
  const int ntb2 = (w >> 2) * 4;
  const _Float16* hRow = sH + (mt * 16 + ln) * 264;
  #pragma unroll
  for (int ks = 0; ks < 8; ++ks) {
    HFrag a;
    const int ak = ks * 32 + hs * 8;
    a.h[0] = *(const h8v*)(hRow + ak);
    a.h[1] = *(const h8v*)(hRow + ak + 16);
    #pragma unroll
    for (int q = 0; q < 4; ++q) {
      const int n = (ntb2 + q) * 16 + ln;
      HFrag bf;
      const _Float16* bp = sW2 + n * 264 + ks * 32 + hs * 8;
      bf.h[0] = *(const h8v*)bp;
      bf.h[1] = *(const h8v*)(bp + 16);
      acc2[q] = __builtin_amdgcn_wmma_f32_16x16x32_f16(
          false, a.v, false, bf.v, (short)0, acc2[q], false, false);
    }
  }

  // ---- epilogue 2: relu(o + b2) * mask -> out (fp32)
  {
    float bbv[4];
    #pragma unroll
    for (int q = 0; q < 4; ++q) bbv[q] = b2[(ntb2 + q) * 16 + ln];
    #pragma unroll
    for (int r = 0; r < 8; ++r) {
      const int m = mt * 16 + r + 8 * hs;
      const size_t prow = (size_t)row0 + m;    // == b*65536 + p, matches flat mask
      const float mk = mask[prow];
      float* orow = out + prow * 128;
      #pragma unroll
      for (int q = 0; q < 4; ++q) {
        const int n = (ntb2 + q) * 16 + ln;
        orow[n] = fmaxf(acc2[q][r] + bbv[q], 0.f) * mk;
      }
    }
  }
}

// ---------------------------------------------------------------------------
extern "C" void kernel_launch(void* const* d_in, const int* in_sizes, int n_in,
                              void* d_out, int out_size, void* d_ws, size_t ws_size,
                              hipStream_t stream) {
  const float* nodes = (const float*)d_in[0];
  const float* edges = (const float*)d_in[1];
  const float* mask  = (const float*)d_in[2];
  const float* W1    = (const float*)d_in[3];
  const float* b1    = (const float*)d_in[4];
  const float* W2    = (const float*)d_in[5];
  const float* b2    = (const float*)d_in[6];
  float* out = (float*)d_out;

  // workspace: pib (1 MB) | pjw (1 MB) | w1et (64 KB f16) | w2t (64 KB f16)
  float* pib = (float*)d_ws;
  float* pjw = pib + 4 * 256 * 256;
  _Float16* w1et = (_Float16*)(pjw + 4 * 256 * 256);
  _Float16* w2t  = w1et + 256 * 128;

  mp_proj<<<4 * 256, 256, 0, stream>>>(nodes, W1, b1, pib, pjw);
  mp_wcvt<<<256, 256, 0, stream>>>(W1, W2, w1et, w2t);
  // 262144 rows / 64 rows per WG = 4096 WGs; 120832 B dynamic LDS
  mp_main<<<4096, 256, 120832, stream>>>(edges, mask, pib, pjw, w1et, w2t, b2, out);
}
